// Attention_25280177504468
// MI455X (gfx1250) — matrix operand order repaired and checked by reading, not compiled, not run on hardware
//
#include <hip/hip_runtime.h>
#include <hip/hip_bf16.h>

// ---------------------------------------------------------------------------
// MI455X (gfx1250) attention block, bf16-WMMA pipeline.
//   - weights/activations pre-converted to bf16 in workspace (L2-resident,
//     ~84 MB vs 168 MB fp32) -> GEMM inner loops load bf16 directly, no VALU
//     conversion on the hot path
//   - v_wmma_f32_16x16x32_bf16 everywhere, fp32 accumulate
//   - fp32 softmax with wave32 half-wave shuffle reductions
//   - native (__bf16) casts (hardware cvt) where conversion is unavoidable
//   - deterministic fallback (B = f32 with on-the-fly cast) if ws too small
// ---------------------------------------------------------------------------

typedef __attribute__((ext_vector_type(16))) __bf16          v16bf;
typedef __attribute__((ext_vector_type(16))) unsigned short  v16u;
typedef __attribute__((ext_vector_type(8)))  float           v8f;

#define DIM   4096
#define NH    32
#define HD    128
#define NKV   8
#define SEQ   128
#define BSZ   4
#define MROWS (BSZ * SEQ)   /* 512  */
#define KVDIM (NKV * HD)    /* 1024 */

union U16 { v16bf bv; uint4 q[2]; };

__device__ __forceinline__ unsigned short bfbits(float f) {
  __bf16 h = (__bf16)f;                       // hardware f32->bf16 (RNE)
  return __builtin_bit_cast(unsigned short, h);
}

__device__ __forceinline__ v16bf bfcast(v16u u) {
  return __builtin_bit_cast(v16bf, u);
}

__device__ __forceinline__ void pack8c(v16bf& v, int base, float4 a, float4 b) {
  v[base + 0] = (__bf16)a.x; v[base + 1] = (__bf16)a.y;
  v[base + 2] = (__bf16)a.z; v[base + 3] = (__bf16)a.w;
  v[base + 4] = (__bf16)b.x; v[base + 5] = (__bf16)b.y;
  v[base + 6] = (__bf16)b.z; v[base + 7] = (__bf16)b.w;
}

// ---------------------------------------------------------------------------
// f32 -> bf16 bulk conversion, 8 elements / thread, vector loads + stores.
// ---------------------------------------------------------------------------
__global__ __launch_bounds__(256)
void cvt_kernel(const float* __restrict__ in, unsigned short* __restrict__ out,
                int n8) {
  int i = blockIdx.x * blockDim.x + threadIdx.x;
  if (i >= n8) return;
  const float4* p = reinterpret_cast<const float4*>(in) + (size_t)i * 2;
  float4 a = p[0], b = p[1];
  union { unsigned short s[8]; uint4 q; } u;
  u.s[0] = bfbits(a.x); u.s[1] = bfbits(a.y);
  u.s[2] = bfbits(a.z); u.s[3] = bfbits(a.w);
  u.s[4] = bfbits(b.x); u.s[5] = bfbits(b.y);
  u.s[6] = bfbits(b.z); u.s[7] = bfbits(b.w);
  reinterpret_cast<uint4*>(out)[i] = u.q;
}

// ---------------------------------------------------------------------------
// C[m,n] = sum_k A[m,k]*B[n,k], A/B bf16 row-major (K-contiguous), C f32.
// grid = (N/64, M/64), block = 128 (4 waves); wave -> 16x64 strip (4 tiles).
// ---------------------------------------------------------------------------
__global__ __launch_bounds__(128)
void gemm_bb(const unsigned short* __restrict__ A,
             const unsigned short* __restrict__ B,
             float* __restrict__ C, int K, int lda, int ldb, int ldc) {
  const int lane  = threadIdx.x & 31;
  const int wave  = threadIdx.x >> 5;
  const int m0    = blockIdx.y * 64 + wave * 16;
  const int n0    = blockIdx.x * 64;
  const int r     = lane & 15;
  const int khalf = lane >> 4;

  v8f zero = {};
  v8f acc[4];
#pragma unroll
  for (int t = 0; t < 4; ++t) acc[t] = zero;

  const unsigned short* arow = A + (size_t)(m0 + r) * lda;

  for (int k = 0; k < K; k += 32) {
    // A tile: elems 0-15 -> K = k + 16*khalf + i
    U16 ua;
    ua.q[0] = *reinterpret_cast<const uint4*>(arow + k + khalf * 16);
    ua.q[1] = *reinterpret_cast<const uint4*>(arow + k + khalf * 16 + 8);
    if (k + 32 < K) __builtin_prefetch(arow + k + 32, 0, 1);
#pragma unroll
    for (int t = 0; t < 4; ++t) {
      // B tile: lane col = n, elems 0-15 -> K = k + 16*khalf + i
      const unsigned short* bcol =
          B + (size_t)(n0 + t * 16 + r) * ldb + k + khalf * 16;
      U16 ub;
      ub.q[0] = reinterpret_cast<const uint4*>(bcol)[0];
      ub.q[1] = reinterpret_cast<const uint4*>(bcol)[1];
      acc[t] = __builtin_amdgcn_wmma_f32_16x16x32_bf16(
          false, ua.bv, false, ub.bv, (short)0, acc[t], false, false);
    }
  }

#pragma unroll
  for (int t = 0; t < 4; ++t) {
    const int col = n0 + t * 16 + r;
#pragma unroll
    for (int rr = 0; rr < 8; ++rr) {
      C[(size_t)(m0 + khalf * 8 + rr) * ldc + col] = acc[t][rr];
    }
  }
}

// ---------------------------------------------------------------------------
// Fallback GEMM: A bf16, B f32 (converted on the fly with hardware casts).
// ---------------------------------------------------------------------------
__global__ __launch_bounds__(128)
void gemm_bf(const unsigned short* __restrict__ A,
             const float* __restrict__ B,
             float* __restrict__ C, int K, int lda, int ldb, int ldc) {
  const int lane  = threadIdx.x & 31;
  const int wave  = threadIdx.x >> 5;
  const int m0    = blockIdx.y * 64 + wave * 16;
  const int n0    = blockIdx.x * 64;
  const int r     = lane & 15;
  const int khalf = lane >> 4;

  v8f zero = {};
  v8f acc[4];
#pragma unroll
  for (int t = 0; t < 4; ++t) acc[t] = zero;

  const unsigned short* arow = A + (size_t)(m0 + r) * lda;

  for (int k = 0; k < K; k += 32) {
    U16 ua;
    ua.q[0] = *reinterpret_cast<const uint4*>(arow + k + khalf * 16);
    ua.q[1] = *reinterpret_cast<const uint4*>(arow + k + khalf * 16 + 8);
    if (k + 32 < K) __builtin_prefetch(arow + k + 32, 0, 1);
#pragma unroll
    for (int t = 0; t < 4; ++t) {
      const float* bcol = B + (size_t)(n0 + t * 16 + r) * ldb + k + khalf * 16;
      const float4* bp = reinterpret_cast<const float4*>(bcol);
      float4 b0 = bp[0], b1 = bp[1], b2 = bp[2], b3 = bp[3];
      v16bf bv;
      pack8c(bv, 0, b0, b1);
      pack8c(bv, 8, b2, b3);
      acc[t] = __builtin_amdgcn_wmma_f32_16x16x32_bf16(
          false, ua.bv, false, bv, (short)0, acc[t], false, false);
    }
  }

#pragma unroll
  for (int t = 0; t < 4; ++t) {
    const int col = n0 + t * 16 + r;
#pragma unroll
    for (int rr = 0; rr < 8; ++rr) {
      C[(size_t)(m0 + khalf * 8 + rr) * ldc + col] = acc[t][rr];
    }
  }
}

// ---------------------------------------------------------------------------
// RoPE over q (512 x 4096, 32 heads) and k (512 x 1024, 8 heads), in place.
// ---------------------------------------------------------------------------
__global__ void rope_kernel(float* __restrict__ q, float* __restrict__ k,
                            const float* __restrict__ cosT,
                            const float* __restrict__ sinT) {
  const int QP = MROWS * (DIM / 2);
  const int KP = MROWS * (KVDIM / 2);
  int idx = blockIdx.x * blockDim.x + threadIdx.x;
  if (idx < QP) {
    int row = idx / (DIM / 2);
    int j   = idx % (DIM / 2);
    int hp  = j & 63;
    int s   = row & (SEQ - 1);
    float c = cosT[s * 64 + hp], sn = sinT[s * 64 + hp];
    float* p = q + (size_t)row * DIM + j * 2;
    float x0 = p[0], x1 = p[1];
    p[0] = x0 * c - x1 * sn;
    p[1] = x0 * sn + x1 * c;
  } else if (idx < QP + KP) {
    int i2  = idx - QP;
    int row = i2 / (KVDIM / 2);
    int j   = i2 % (KVDIM / 2);
    int hp  = j & 63;
    int s   = row & (SEQ - 1);
    float c = cosT[s * 64 + hp], sn = sinT[s * 64 + hp];
    float* p = k + (size_t)row * KVDIM + j * 2;
    float x0 = p[0], x1 = p[1];
    p[0] = x0 * c - x1 * sn;
    p[1] = x0 * sn + x1 * c;
  }
}

// ---------------------------------------------------------------------------
// Attention core: one block (256 thr = 8 waves) per (head, batch).
// S = QK^T*scale + causal -> fp32 softmax -> O = P*V. P transits LDS bf16 to
// fix C-layout -> A-layout; V^T staged in LDS. Output written as bf16 so the
// output projection streams bf16.
// ---------------------------------------------------------------------------
__global__ __launch_bounds__(256)
void attn_kernel(const float* __restrict__ q, const float* __restrict__ kk,
                 const float* __restrict__ vv, unsigned short* __restrict__ o) {
  __shared__ unsigned short vt[HD][SEQ];       // V^T bf16 bits: vt[d][j]
  __shared__ unsigned short pls[8][16][SEQ];   // per-wave P strips

  const int h     = blockIdx.x;
  const int b     = blockIdx.y;
  const int kvh   = h >> 2;
  const int tid   = threadIdx.x;
  const int lane  = tid & 31;
  const int wave  = tid >> 5;
  const int r     = lane & 15;
  const int khalf = lane >> 4;

  for (int it = 0; it < (SEQ * HD) / 256; ++it) {
    int idx = it * 256 + tid;
    int j = idx >> 7;
    int d = idx & 127;
    vt[d][j] = bfbits(vv[(size_t)(b * SEQ + j) * KVDIM + kvh * HD + d]);
  }
  __syncthreads();

  const int qrow0 = wave * 16;
  const float* qrow = q + (size_t)(b * SEQ + qrow0 + r) * DIM + h * HD;

  v8f zero = {};
  v8f acc[8];
#pragma unroll
  for (int t = 0; t < 8; ++t) acc[t] = zero;

  // ---- S = Q K^T (16x128 strip per wave), K-dim = head_dim
#pragma unroll
  for (int kd = 0; kd < HD; kd += 32) {
    const float4* ap0 = reinterpret_cast<const float4*>(qrow + kd + khalf * 16);
    const float4* ap1 = reinterpret_cast<const float4*>(qrow + kd + khalf * 16 + 8);
    float4 a0 = ap0[0], a1 = ap0[1], a2 = ap1[0], a3 = ap1[1];
    v16bf av;
    pack8c(av, 0, a0, a1);
    pack8c(av, 8, a2, a3);
#pragma unroll
    for (int t = 0; t < 8; ++t) {
      int j = t * 16 + r;
      const float4* bp = reinterpret_cast<const float4*>(
          kk + (size_t)(b * SEQ + j) * KVDIM + kvh * HD + kd + khalf * 16);
      float4 b0 = bp[0], b1 = bp[1], b2 = bp[2], b3 = bp[3];
      v16bf bv;
      pack8c(bv, 0, b0, b1);
      pack8c(bv, 8, b2, b3);
      acc[t] = __builtin_amdgcn_wmma_f32_16x16x32_bf16(
          false, av, false, bv, (short)0, acc[t], false, false);
    }
  }

  // ---- scale + causal mask + fp32 softmax
  const float scale = 0.08838834764831845f;    // 1/sqrt(128)
#pragma unroll
  for (int rr = 0; rr < 8; ++rr) {
    const int row = qrow0 + khalf * 8 + rr;
    float m = -3.4e38f;
#pragma unroll
    for (int t = 0; t < 8; ++t) {
      int col = t * 16 + r;
      float v = acc[t][rr] * scale + ((col > row) ? -1.0e9f : 0.0f);
      acc[t][rr] = v;
      m = fmaxf(m, v);
    }
    for (int mk = 1; mk < 16; mk <<= 1) m = fmaxf(m, __shfl_xor(m, mk, 32));
    float s = 0.0f;
#pragma unroll
    for (int t = 0; t < 8; ++t) {
      float e = __expf(acc[t][rr] - m);
      acc[t][rr] = e;
      s += e;
    }
    for (int mk = 1; mk < 16; mk <<= 1) s += __shfl_xor(s, mk, 32);
    float inv = 1.0f / s;
#pragma unroll
    for (int t = 0; t < 8; ++t) {
      pls[wave][khalf * 8 + rr][t * 16 + r] = bfbits(acc[t][rr] * inv);
    }
  }
  __syncthreads();

  // ---- O = P V (K-dim = seq); A from LDS, B from LDS V^T
  v8f oacc[8];
#pragma unroll
  for (int t = 0; t < 8; ++t) oacc[t] = zero;

#pragma unroll
  for (int jk = 0; jk < SEQ; jk += 32) {
    v16u au;
#pragma unroll
    for (int i = 0; i < 8; ++i) au[i]     = pls[wave][r][jk + khalf * 16 + i];
#pragma unroll
    for (int i = 0; i < 8; ++i) au[8 + i] = pls[wave][r][jk + khalf * 16 + 8 + i];
    v16bf av = bfcast(au);
#pragma unroll
    for (int t = 0; t < 8; ++t) {
      int d = t * 16 + r;
      v16u bu;
#pragma unroll
      for (int i = 0; i < 16; ++i) bu[i] = vt[d][jk + khalf * 16 + i];
      v16bf bv = bfcast(bu);
      oacc[t] = __builtin_amdgcn_wmma_f32_16x16x32_bf16(
          false, av, false, bv, (short)0, oacc[t], false, false);
    }
  }

  // ---- store O strip as bf16 into (b, s, h*HD + d)
#pragma unroll
  for (int t = 0; t < 8; ++t) {
    int d = t * 16 + r;
#pragma unroll
    for (int rr = 0; rr < 8; ++rr) {
      int row = qrow0 + khalf * 8 + rr;
      o[(size_t)(b * SEQ + row) * DIM + h * HD + d] = bfbits(oacc[t][rr]);
    }
  }
}

// ---------------------------------------------------------------------------
extern "C" void kernel_launch(void* const* d_in, const int* in_sizes, int n_in,
                              void* d_out, int out_size, void* d_ws, size_t ws_size,
                              hipStream_t stream) {
  (void)in_sizes; (void)n_in; (void)out_size;
  const float* x  = (const float*)d_in[0];
  const float* wq = (const float*)d_in[1];
  const float* wk = (const float*)d_in[2];
  const float* wv = (const float*)d_in[3];
  const float* wo = (const float*)d_in[4];
  const float* fc = (const float*)d_in[7];   // freqs_cos
  const float* fs = (const float*)d_in[8];   // freqs_sin
  float* out = (float*)d_out;

  const size_t XE  = (size_t)MROWS * DIM;     // 2,097,152 elems
  const size_t KVE = (size_t)MROWS * KVDIM;   //   524,288 elems
  const size_t WQE = (size_t)DIM * DIM;       // 16,777,216 elems
  const size_t WKE = (size_t)KVDIM * DIM;     //  4,194,304 elems

  // ---- common workspace (20 MB): xbf | abf | qws | kws | vws
  char* w = (char*)d_ws;
  unsigned short* xbf = (unsigned short*)w; w += XE * 2;
  unsigned short* abf = (unsigned short*)w; w += XE * 2;
  float* qws = (float*)w; w += XE * 4;
  float* kws = (float*)w; w += KVE * 4;
  float* vws = (float*)w; w += KVE * 4;
  const size_t base_need = (size_t)(w - (char*)d_ws);
  const size_t big_need  = base_need + (WQE + WKE + WKE + WQE) * 2;  // ~104 MB
  const bool big = (ws_size >= big_need);

  // x -> bf16 (both paths)
  cvt_kernel<<<(int)((XE / 8 + 255) / 256), 256, 0, stream>>>(x, xbf, (int)(XE / 8));

  if (big) {
    unsigned short* wqbf = (unsigned short*)w; w += WQE * 2;
    unsigned short* wkbf = (unsigned short*)w; w += WKE * 2;
    unsigned short* wvbf = (unsigned short*)w; w += WKE * 2;
    unsigned short* wobf = (unsigned short*)w;

    cvt_kernel<<<(int)((WQE / 8 + 255) / 256), 256, 0, stream>>>(wq, wqbf, (int)(WQE / 8));
    cvt_kernel<<<(int)((WKE / 8 + 255) / 256), 256, 0, stream>>>(wk, wkbf, (int)(WKE / 8));
    cvt_kernel<<<(int)((WKE / 8 + 255) / 256), 256, 0, stream>>>(wv, wvbf, (int)(WKE / 8));
    cvt_kernel<<<(int)((WQE / 8 + 255) / 256), 256, 0, stream>>>(wo, wobf, (int)(WQE / 8));

    gemm_bb<<<dim3(DIM / 64, MROWS / 64), 128, 0, stream>>>(xbf, wqbf, qws, DIM, DIM, DIM, DIM);
    gemm_bb<<<dim3(KVDIM / 64, MROWS / 64), 128, 0, stream>>>(xbf, wkbf, kws, DIM, DIM, DIM, KVDIM);
    gemm_bb<<<dim3(KVDIM / 64, MROWS / 64), 128, 0, stream>>>(xbf, wvbf, vws, DIM, DIM, DIM, KVDIM);

    const int totalPairs = MROWS * (DIM / 2) + MROWS * (KVDIM / 2);
    rope_kernel<<<(totalPairs + 255) / 256, 256, 0, stream>>>(qws, kws, fc, fs);

    attn_kernel<<<dim3(NH, BSZ), 256, 0, stream>>>(qws, kws, vws, abf);

    gemm_bb<<<dim3(DIM / 64, MROWS / 64), 128, 0, stream>>>(abf, wobf, out, DIM, DIM, DIM, DIM);
  } else {
    gemm_bf<<<dim3(DIM / 64, MROWS / 64), 128, 0, stream>>>(xbf, wq, qws, DIM, DIM, DIM, DIM);
    gemm_bf<<<dim3(KVDIM / 64, MROWS / 64), 128, 0, stream>>>(xbf, wk, kws, DIM, DIM, DIM, KVDIM);
    gemm_bf<<<dim3(KVDIM / 64, MROWS / 64), 128, 0, stream>>>(xbf, wv, vws, DIM, DIM, DIM, KVDIM);

    const int totalPairs = MROWS * (DIM / 2) + MROWS * (KVDIM / 2);
    rope_kernel<<<(totalPairs + 255) / 256, 256, 0, stream>>>(qws, kws, fc, fs);

    attn_kernel<<<dim3(NH, BSZ), 256, 0, stream>>>(qws, kws, vws, abf);

    gemm_bf<<<dim3(DIM / 64, MROWS / 64), 128, 0, stream>>>(abf, wo, out, DIM, DIM, DIM, DIM);
  }
}
